// GeoCGNN_62637803045234
// MI455X (gfx1250) — compile-verified, matrix operand order checked by reading, not run on hardware
//
#include <hip/hip_runtime.h>
#include <hip/hip_bf16.h>

#define HDIM 128
#define KC 64
#define NEDGES 160000
#define NNODES 10000
#define NGRAPH 100
#define ETILE 32

typedef __attribute__((ext_vector_type(16))) _Float16 v16h;
typedef __attribute__((ext_vector_type(8)))  _Float16 v8h;
typedef __attribute__((ext_vector_type(8)))  float    v8f;

// Load a 16x32 f16 WMMA A/B tile fragment for this lane.
// Caller passes pointer already offset to: row_base + kbase + ko,
// where ko = (lane<16) ? 0 : 8.  Elements [0..7] from p, [8..15] from p+16.
__device__ __forceinline__ v16h ld_ab_tile(const _Float16* p) {
    v8h lo = *(const v8h*)(p);
    v8h hi = *(const v8h*)(p + 16);
    v16h r;
#pragma unroll
    for (int i = 0; i < 8; ++i) { r[i] = lo[i]; r[i + 8] = hi[i]; }
    return r;
}

__device__ __forceinline__ v8f wmma16(v16h a, v16h b, v8f c) {
    return __builtin_amdgcn_wmma_f32_16x16x32_f16(
        false, a, false, b, (short)0, c, false, false);
}

__device__ __forceinline__ float fast_rcp(float x) {
    return __builtin_amdgcn_rcpf(x);   // v_rcp_f32, ~1 ULP
}
__device__ __forceinline__ float sigmoidf_(float x) {
    return fast_rcp(1.0f + __expf(-x));
}
__device__ __forceinline__ float eluf_(float x) {
    return x > 0.0f ? x : (__expf(x) - 1.0f);
}

// ---------------- utility kernels ----------------

__global__ void zero_kernel(float* p, int n) {
    int i = blockIdx.x * blockDim.x + threadIdx.x;
    if (i < n) p[i] = 0.0f;
}

// dst[n*K + k] = (f16) src[k*N + n]    (transpose [K][N] f32 -> [N][K] f16)
__global__ void convert_transpose_kernel(const float* __restrict__ src,
                                         _Float16* __restrict__ dst,
                                         int K, int N) {
    int idx = blockIdx.x * blockDim.x + threadIdx.x;
    if (idx < K * N) {
        int k = idx / N, n = idx % N;
        dst[n * K + k] = (_Float16)src[idx];
    }
}

// ---------------- embedding: h = sigmoid(x @ emb_W) ----------------
__global__ __launch_bounds__(256)
void emb_kernel(const float* __restrict__ x,
                const _Float16* __restrict__ Wt,   // [128][128]
                float* __restrict__ hf32,
                _Float16* __restrict__ hhalf) {
    __shared__ __align__(16) _Float16 aT[16 * HDIM];
    const int tid  = threadIdx.x;
    const int lane = tid & 31;
    const int wave = tid >> 5;
    const int nbase = blockIdx.x * 16;

    {
        int rid = tid >> 4, seg = tid & 15;
        const float* s = x + (size_t)(nbase + rid) * HDIM + seg * 8;
#pragma unroll
        for (int j = 0; j < 8; ++j) aT[rid * HDIM + seg * 8 + j] = (_Float16)s[j];
    }
    __syncthreads();

    const int nb   = wave * 16;
    const int mrow = lane & 15;
    const int ko   = (lane < 16) ? 0 : 8;
    v8f acc = {};
#pragma unroll
    for (int kt = 0; kt < 4; ++kt) {
        int kb = kt * 32 + ko;
        v16h a = ld_ab_tile(&aT[mrow * HDIM + kb]);
        v16h b = ld_ab_tile(&Wt[(size_t)(nb + mrow) * HDIM + kb]);
        acc = wmma16(a, b, acc);
    }
#pragma unroll
    for (int v = 0; v < 8; ++v) {
        int row = v + 8 * (lane >> 4);
        int col = nb + mrow;
        float h = sigmoidf_(acc[v]);
        size_t o = (size_t)(nbase + row) * HDIM + col;
        hf32[o]  = h;
        hhalf[o] = (_Float16)h;
    }
}

// ---------------- edge convolution + scatter ----------------
// grid: NEDGES/32 workgroups of 256 (8 waves); tile = 32 edges (two M tiles).
// Each wave owns 16 output columns and computes BOTH M tiles, reusing every
// B fragment for 2 WMMAs (doubles matrix-op density per global load).
__global__ __launch_bounds__(256)
void edge_conv_kernel(const float* __restrict__ edist,
                      const float* __restrict__ csets,   // [E][64] f32
                      const float* __restrict__ pwave,   // [E][64] f32
                      const int* __restrict__ esrc,
                      const int* __restrict__ etgt,
                      const _Float16* __restrict__ hhalf,
                      float* __restrict__ hacc,          // f32 h, scatter target
                      const _Float16* __restrict__ Wgate_t, // [128][384]
                      const _Float16* __restrict__ Wmlp_t,  // [128][384]
                      const _Float16* __restrict__ W1_t,    // [128][64]
                      const _Float16* __restrict__ W2g_t,   // [64][64]
                      const _Float16* __restrict__ W2_t) {  // [128][64]
    __shared__ __align__(16) _Float16 feA[ETILE * 384];   // 24 KB
    __shared__ __align__(16) _Float16 csA[ETILE * KC];    //  4 KB
    __shared__ __align__(16) _Float16 pwA[ETILE * KC];
    __shared__ __align__(16) _Float16 pwmA[ETILE * KC];
    __shared__ float maskS[ETILE];
    __shared__ int   srcS[ETILE];

    const int tid   = threadIdx.x;
    const int lane  = tid & 31;
    const int wave  = tid >> 5;
    const int ebase = blockIdx.x * ETILE;

    if (tid < ETILE) {
        float r = edist[ebase + tid];
        maskS[tid] = (r < 8.0f) ? 1.0f : 0.0f;
        srcS[tid]  = esrc[ebase + tid];
    }

    // stage fe = [ni | nj | (ni - nj)/r]  (16 threads per edge, 2 passes)
#pragma unroll
    for (int it = 0; it < 2; ++it) {
        int t = tid + it * 256;
        int eidx = t >> 4, seg = t & 15;
        int e = ebase + eidx;
        float rinv = fast_rcp(edist[e]);
        const _Float16* pi = hhalf + (size_t)esrc[e] * HDIM + seg * 8;
        const _Float16* pj = hhalf + (size_t)etgt[e] * HDIM + seg * 8;
        v8h ni = *(const v8h*)pi;
        v8h nj = *(const v8h*)pj;
        v8h di;
#pragma unroll
        for (int j = 0; j < 8; ++j)
            di[j] = (_Float16)(((float)ni[j] - (float)nj[j]) * rinv);
        *(v8h*)&feA[eidx * 384 + seg * 8]       = ni;
        *(v8h*)&feA[eidx * 384 + 128 + seg * 8] = nj;
        *(v8h*)&feA[eidx * 384 + 256 + seg * 8] = di;
    }
    // stage combine_sets / plane_wave (f32 -> f16), 2 passes
#pragma unroll
    for (int it = 0; it < 2; ++it) {
        int t = tid + it * 256;
        int eidx = t >> 4, q = t & 15;
        int e = ebase + eidx;
        float4 c = *(const float4*)(csets + (size_t)e * KC + q * 4);
        float4 p = *(const float4*)(pwave + (size_t)e * KC + q * 4);
        csA[eidx * KC + q * 4 + 0] = (_Float16)c.x;
        csA[eidx * KC + q * 4 + 1] = (_Float16)c.y;
        csA[eidx * KC + q * 4 + 2] = (_Float16)c.z;
        csA[eidx * KC + q * 4 + 3] = (_Float16)c.w;
        pwA[eidx * KC + q * 4 + 0] = (_Float16)p.x;
        pwA[eidx * KC + q * 4 + 1] = (_Float16)p.y;
        pwA[eidx * KC + q * 4 + 2] = (_Float16)p.z;
        pwA[eidx * KC + q * 4 + 3] = (_Float16)p.w;
    }
    __syncthreads();

    const int mrow = lane & 15;
    const int ko   = (lane < 16) ? 0 : 8;

    // phase A: pwm = plane_wave * sigmoid(plane_wave @ W2g)
    // 8 tiles: wave -> (mtile = wave>>2, ntile = wave&3)
    {
        const int mt = (wave >> 2) * 16;
        const int nb = (wave & 3) * 16;
        v8f acc = {};
#pragma unroll
        for (int kt = 0; kt < 2; ++kt) {
            int kb = kt * 32 + ko;
            v16h a = ld_ab_tile(&pwA[(mt + mrow) * KC + kb]);
            v16h b = ld_ab_tile(&W2g_t[(size_t)(nb + mrow) * KC + kb]);
            acc = wmma16(a, b, acc);
        }
#pragma unroll
        for (int v = 0; v < 8; ++v) {
            int row = mt + v + 8 * (lane >> 4);
            int col = nb + mrow;
            float pwv = (float)pwA[row * KC + col];
            pwmA[row * KC + col] = (_Float16)(pwv * sigmoidf_(acc[v]));
        }
    }
    __syncthreads();

    // phase B: each wave owns 16 output columns, both M tiles
    const int nb = wave * 16;
    v8f g0 = {}, g1 = {}, m0 = {}, m1 = {}, z0 = {}, z1 = {};
    const _Float16* gR = Wgate_t + (size_t)(nb + mrow) * 384;
    const _Float16* mR = Wmlp_t  + (size_t)(nb + mrow) * 384;
#pragma unroll
    for (int kt = 0; kt < 12; ++kt) {
        int kb = kt * 32 + ko;
        v16h a0 = ld_ab_tile(&feA[mrow * 384 + kb]);
        v16h a1 = ld_ab_tile(&feA[(16 + mrow) * 384 + kb]);
        v16h bg = ld_ab_tile(gR + kb);
        v16h bm = ld_ab_tile(mR + kb);
        g0 = wmma16(a0, bg, g0);
        g1 = wmma16(a1, bg, g1);
        m0 = wmma16(a0, bm, m0);
        m1 = wmma16(a1, bm, m1);
    }
#pragma unroll
    for (int kt = 0; kt < 2; ++kt) {   // z1 = combine_sets @ W1
        int kb = kt * 32 + ko;
        v16h b  = ld_ab_tile(&W1_t[(size_t)(nb + mrow) * KC + kb]);
        z0 = wmma16(ld_ab_tile(&csA[mrow * KC + kb]), b, z0);
        z1 = wmma16(ld_ab_tile(&csA[(16 + mrow) * KC + kb]), b, z1);
    }
#pragma unroll
    for (int kt = 0; kt < 2; ++kt) {   // + z2 = pwm @ W2
        int kb = kt * 32 + ko;
        v16h b  = ld_ab_tile(&W2_t[(size_t)(nb + mrow) * KC + kb]);
        z0 = wmma16(ld_ab_tile(&pwmA[mrow * KC + kb]), b, z0);
        z1 = wmma16(ld_ab_tile(&pwmA[(16 + mrow) * KC + kb]), b, z1);
    }
    // epilogue: z = sig(gate)*elu(mlp)*(z1+z2)*mask; scatter-add to h
#pragma unroll
    for (int v = 0; v < 8; ++v) {
        int row = v + 8 * (lane >> 4);
        int col = nb + mrow;
        float za = sigmoidf_(g0[v]) * eluf_(m0[v]) * z0[v] * maskS[row];
        atomicAdd(&hacc[(size_t)srcS[row] * HDIM + col], za);
        float zb = sigmoidf_(g1[v]) * eluf_(m1[v]) * z1[v] * maskS[16 + row];
        atomicAdd(&hacc[(size_t)srcS[16 + row] * HDIM + col], zb);
    }
}

// ---------------- node update: pool + psi ----------------
__global__ __launch_bounds__(256)
void node_kernel(float* __restrict__ hf32,
                 _Float16* __restrict__ hhalf,
                 float* __restrict__ pooled,        // [G][128]
                 const int* __restrict__ gidx,
                 const _Float16* __restrict__ P1t,  // [128][128]
                 const _Float16* __restrict__ P2t,
                 const _Float16* __restrict__ Pst) {
    __shared__ __align__(16) _Float16 hA[16 * HDIM];
    __shared__ int gS[16];
    const int tid   = threadIdx.x;
    const int lane  = tid & 31;
    const int wave  = tid >> 5;
    const int nbase = blockIdx.x * 16;

    if (tid < 16) gS[tid] = gidx[nbase + tid];
    {
        int rid = tid >> 4, seg = tid & 15;
        const float* s = hf32 + (size_t)(nbase + rid) * HDIM + seg * 8;
#pragma unroll
        for (int j = 0; j < 8; ++j) hA[rid * HDIM + seg * 8 + j] = (_Float16)s[j];
    }
    __syncthreads();

    const int nb   = wave * 16;
    const int mrow = lane & 15;
    const int ko   = (lane < 16) ? 0 : 8;
    v8f a1 = {}, a2 = {}, ap = {};
#pragma unroll
    for (int kt = 0; kt < 4; ++kt) {
        int kb = kt * 32 + ko;
        v16h a = ld_ab_tile(&hA[mrow * HDIM + kb]);
        a1 = wmma16(a, ld_ab_tile(&P1t[(size_t)(nb + mrow) * HDIM + kb]), a1);
        a2 = wmma16(a, ld_ab_tile(&P2t[(size_t)(nb + mrow) * HDIM + kb]), a2);
        ap = wmma16(a, ld_ab_tile(&Pst[(size_t)(nb + mrow) * HDIM + kb]), ap);
    }
#pragma unroll
    for (int v = 0; v < 8; ++v) {
        int row  = v + 8 * (lane >> 4);
        int col  = nb + mrow;
        int node = nbase + row;
        float zp = eluf_(a1[v] * a2[v]);
        atomicAdd(&pooled[(size_t)gS[row] * HDIM + col], zp);
        float hn = eluf_(ap[v]);
        size_t o = (size_t)node * HDIM + col;
        hf32[o]  = hn;
        hhalf[o] = (_Float16)hn;
    }
}

// ---------------- readout: y = elu(elu(pooled @ lr1) @ lr2) ----------------
__global__ __launch_bounds__(256)
void readout_kernel(const float* __restrict__ pooled,
                    const _Float16* __restrict__ L1t,  // [64][128]
                    const _Float16* __restrict__ L2t,  // [32][64]
                    float* __restrict__ out) {
    __shared__ __align__(16) _Float16 pA[16 * HDIM];
    __shared__ __align__(16) _Float16 y1A[16 * 64];
    const int tid   = threadIdx.x;
    const int lane  = tid & 31;
    const int wave  = tid >> 5;
    const int rbase = blockIdx.x * 16;

    {
        int rid = tid >> 4, seg = tid & 15;
        int row = rbase + rid;
#pragma unroll
        for (int j = 0; j < 8; ++j) {
            float v = (row < NGRAPH) ? pooled[(size_t)row * HDIM + seg * 8 + j] : 0.0f;
            pA[rid * HDIM + seg * 8 + j] = (_Float16)v;
        }
    }
    __syncthreads();

    const int mrow = lane & 15;
    const int ko   = (lane < 16) ? 0 : 8;

    if (wave < 4) {   // y1 = elu(pooled @ lr1): 64 cols
        const int nb = wave * 16;
        v8f acc = {};
#pragma unroll
        for (int kt = 0; kt < 4; ++kt) {
            int kb = kt * 32 + ko;
            v16h a = ld_ab_tile(&pA[mrow * HDIM + kb]);
            v16h b = ld_ab_tile(&L1t[(size_t)(nb + mrow) * HDIM + kb]);
            acc = wmma16(a, b, acc);
        }
#pragma unroll
        for (int v = 0; v < 8; ++v) {
            int row = v + 8 * (lane >> 4);
            int col = nb + mrow;
            y1A[row * 64 + col] = (_Float16)eluf_(acc[v]);
        }
    }
    __syncthreads();

    if (wave < 2) {   // y = elu(y1 @ lr2): 32 cols
        const int nb = wave * 16;
        v8f acc = {};
#pragma unroll
        for (int kt = 0; kt < 2; ++kt) {
            int kb = kt * 32 + ko;
            v16h a = ld_ab_tile(&y1A[mrow * 64 + kb]);
            v16h b = ld_ab_tile(&L2t[(size_t)(nb + mrow) * 64 + kb]);
            acc = wmma16(a, b, acc);
        }
#pragma unroll
        for (int v = 0; v < 8; ++v) {
            int row  = v + 8 * (lane >> 4);
            int grow = rbase + row;
            if (grow < NGRAPH) out[(size_t)grow * 32 + nb + mrow] = eluf_(acc[v]);
        }
    }
}

// ---------------- host launcher ----------------
extern "C" void kernel_launch(void* const* d_in, const int* in_sizes, int n_in,
                              void* d_out, int out_size, void* d_ws, size_t ws_size,
                              hipStream_t stream) {
    const float* x     = (const float*)d_in[0];
    const float* edist = (const float*)d_in[1];
    const float* cs    = (const float*)d_in[2];
    const float* pw    = (const float*)d_in[3];
    const float* embW  = (const float*)d_in[4];
    const float* Wgate = (const float*)d_in[5];
    const float* Wmlp  = (const float*)d_in[6];
    const float* W1    = (const float*)d_in[7];
    const float* W2g   = (const float*)d_in[8];
    const float* W2    = (const float*)d_in[9];
    const float* P1    = (const float*)d_in[10];
    const float* P2    = (const float*)d_in[11];
    const float* Psi   = (const float*)d_in[12];
    const float* L1    = (const float*)d_in[13];
    const float* L2    = (const float*)d_in[14];
    const int* esrc    = (const int*)d_in[15];
    const int* etgt    = (const int*)d_in[16];
    const int* gidx    = (const int*)d_in[17];
    float* out         = (float*)d_out;

    char* ws = (char*)d_ws;
    float*     hf32   = (float*)(ws);                         // 5,120,000 B
    _Float16*  hhalf  = (_Float16*)(ws + 5120000);            // 2,560,000 B
    float*     pooled = (float*)(ws + 7680000);               //    51,200 B
    _Float16*  wb     = (_Float16*)(ws + 7731200);            // f16 weights
    _Float16* emb_t  = wb;
    _Float16* gate_t = emb_t  + 16384;
    _Float16* mlp_t  = gate_t + 147456;
    _Float16* W1_t   = mlp_t  + 147456;
    _Float16* W2g_t  = W1_t   + 24576;
    _Float16* W2_t   = W2g_t  + 12288;
    _Float16* P1_t   = W2_t   + 24576;
    _Float16* P2_t   = P1_t   + 49152;
    _Float16* Psi_t  = P2_t   + 49152;
    _Float16* L1_t   = Psi_t  + 49152;
    _Float16* L2_t   = L1_t   + 8192;

    auto ct = [&](const float* s, _Float16* d, int K, int N) {
        int n = K * N;
        convert_transpose_kernel<<<(n + 255) / 256, 256, 0, stream>>>(s, d, K, N);
    };

    zero_kernel<<<(NGRAPH * HDIM + 255) / 256, 256, 0, stream>>>(pooled, NGRAPH * HDIM);

    ct(embW, emb_t, 128, 128);
    for (int i = 0; i < 3; ++i) {
        ct(Wgate + (size_t)i * 384 * 128, gate_t + (size_t)i * 128 * 384, 384, 128);
        ct(Wmlp  + (size_t)i * 384 * 128, mlp_t  + (size_t)i * 128 * 384, 384, 128);
        ct(W1    + (size_t)i * 64 * 128,  W1_t   + (size_t)i * 128 * 64,  64, 128);
        ct(W2g   + (size_t)i * 64 * 64,   W2g_t  + (size_t)i * 64 * 64,   64, 64);
        ct(W2    + (size_t)i * 64 * 128,  W2_t   + (size_t)i * 128 * 64,  64, 128);
        ct(P1    + (size_t)i * 128 * 128, P1_t   + (size_t)i * 128 * 128, 128, 128);
        ct(P2    + (size_t)i * 128 * 128, P2_t   + (size_t)i * 128 * 128, 128, 128);
        ct(Psi   + (size_t)i * 128 * 128, Psi_t  + (size_t)i * 128 * 128, 128, 128);
    }
    ct(L1, L1_t, 128, 64);
    ct(L2, L2_t, 64, 32);

    emb_kernel<<<NNODES / 16, 256, 0, stream>>>(x, emb_t, hf32, hhalf);

    for (int i = 0; i < 3; ++i) {
        edge_conv_kernel<<<NEDGES / ETILE, 256, 0, stream>>>(
            edist, cs, pw, esrc, etgt, hhalf, hf32,
            gate_t + (size_t)i * 128 * 384,
            mlp_t  + (size_t)i * 128 * 384,
            W1_t   + (size_t)i * 128 * 64,
            W2g_t  + (size_t)i * 64 * 64,
            W2_t   + (size_t)i * 128 * 64);
        node_kernel<<<NNODES / 16, 256, 0, stream>>>(
            hf32, hhalf, pooled, gidx,
            P1_t  + (size_t)i * 128 * 128,
            P2_t  + (size_t)i * 128 * 128,
            Psi_t + (size_t)i * 128 * 128);
    }

    readout_kernel<<<(NGRAPH + 15) / 16, 256, 0, stream>>>(pooled, L1_t, L2_t, out);
}
